// FlashAttention_50964081934679
// MI455X (gfx1250) — matrix operand order repaired and checked by reading, not compiled, not run on hardware
//
#include <hip/hip_runtime.h>
#include <hip/hip_bf16.h>

// ---------------------------------------------------------------------------
// CDNA5 / gfx1250 block-attention, f16 WMMA with f32 accumulation.
// Shapes fixed by the reference: B=2, S=2048, D=2048, H=16, hd=128, block=64.
// ---------------------------------------------------------------------------

typedef __attribute__((ext_vector_type(16))) _Float16 v16h;
typedef __attribute__((ext_vector_type(8)))  float    v8f;
typedef __attribute__((ext_vector_type(4)))  _Float16 h4;

#define WMMA_F16(a, b, c) \
  __builtin_amdgcn_wmma_f32_16x16x32_f16(false, (a), false, (b), (short)0, (c), false, false)

// Async global->LDS (GLOBAL_LOAD_ASYNC_TO_LDS_B128, ASYNCcnt) if the
// toolchain exposes it; otherwise fall back to load+ds_store staging.
#if defined(__has_builtin)
#if __has_builtin(__builtin_amdgcn_global_load_async_to_lds_b128) && \
    __has_builtin(__builtin_amdgcn_s_wait_asynccnt)
#define HAVE_ASYNC_LDS 1
#endif
#endif
#ifndef HAVE_ASYNC_LDS
#define HAVE_ASYNC_LDS 0
#endif

#if HAVE_ASYNC_LDS
// Builtin expects pointers to 16-byte int vectors (per hipcc diagnostic):
//   arg0: addrspace(1) int4*  (global source)
//   arg1: addrspace(3) int4*  (LDS destination)
typedef int b128_t __attribute__((vector_size(16)));
typedef __attribute__((address_space(1))) b128_t as1_b128;
typedef __attribute__((address_space(3))) b128_t as3_b128;
#endif

constexpr int B_ = 2, S_ = 2048, D_ = 2048, H_ = 16, HD = 128, NB = 32;

// ---- Fragment loaders (wave32 layouts per CDNA5 ISA §7.12.2) ----

// A-matrix 16x32 f16 (M x K), source row-major, K contiguous, row stride ld.
__device__ __forceinline__ v16h load_a16(const _Float16* p, int ld, int lane) {
  const _Float16* row = p + (size_t)(lane & 15) * ld + ((lane >> 4) << 3);
  v16h a;
#pragma unroll
  for (int pp = 0; pp < 8; ++pp) {
    int k = ((pp & 4) << 2) + ((pp & 3) << 1);  // 0,2,4,6,16,18,20,22
    a[2 * pp]     = row[k];
    a[2 * pp + 1] = row[k + 1];
  }
  return a;
}

// B-matrix 32x16 f16 (K x N) from an N-major / K-contiguous source
// (source stored [N][K] row-major).  lane: N = lane&15, K = 8*(lane>>4)+e (e<8), 16+8*(lane>>4)+(e-8) (e>=8).
__device__ __forceinline__ v16h load_bt16(const _Float16* p, int ld, int lane) {
  const _Float16* row = p + (size_t)(lane & 15) * ld + ((lane >> 4) << 3);
  v16h b;
#pragma unroll
  for (int e = 0; e < 16; ++e) b[e] = row[e < 8 ? e : e + 8];
  return b;
}

// ---------------------------------------------------------------------------
// K0: f32 -> f16 elementwise conversion (grid-strided, float4 granularity)
// ---------------------------------------------------------------------------
__global__ void cvt_f32_f16(const float* __restrict__ src, _Float16* __restrict__ dst,
                            size_t n) {
  size_t i = ((size_t)blockIdx.x * blockDim.x + threadIdx.x) * 4;
  size_t stride = (size_t)gridDim.x * blockDim.x * 4;
  for (; i < n; i += stride) {
    float4 f = *reinterpret_cast<const float4*>(src + i);
    h4 o;
    o[0] = (_Float16)f.x; o[1] = (_Float16)f.y;
    o[2] = (_Float16)f.z; o[3] = (_Float16)f.w;
    *reinterpret_cast<h4*>(dst + i) = o;
  }
}

// ---------------------------------------------------------------------------
// K1/K3: C[M,N] = A[M,K] @ Bt[N,K]^T   (nn.Linear: y = x @ W.T)
// 8 waves/WG, tile M=128 (16 rows/wave) x N=64 (4 WMMA cols/wave), K step 32.
// The shared 64x32 B-tile is staged into LDS once per K-step (async DMA when
// available, double-buffered so the copy of step k+1 overlaps WMMAs of step k).
// MODE 0: store f16 head-major [B,H,S,hd]   (Q, K projections)
// MODE 1: store f16 transposed [B,H,hd,S]   (V projection)
// MODE 2: store f32 row-major [M,N]         (output projection)
// ---------------------------------------------------------------------------
#define BPAD 40  // 32-half rows padded to 40 halfs (80 B): 20n mod 64 is conflict-free

template <int MODE>
__global__ __launch_bounds__(256) void gemm_rt(const _Float16* __restrict__ A,
                                               const _Float16* __restrict__ Bt,
                                               void* __restrict__ Cout,
                                               int M, int N, int K) {
  const int lane = threadIdx.x & 31;
  const int wave = threadIdx.x >> 5;
  const int m0 = blockIdx.x * 128 + wave * 16;
  const int n0 = blockIdx.y * 64;

  __shared__ _Float16 s_b[2][64][BPAD];  // double-buffered B tile (10 KB)

  // staging assignment: thread t copies 16 B: row n = t>>2, chunk cq = t&3
  const int nrow = threadIdx.x >> 2;
  const int cq   = threadIdx.x & 3;
  const _Float16* gsrc_row = Bt + (size_t)(n0 + nrow) * K + cq * 8;

  v8f acc[4] = {};
  const _Float16* a_ptr = A + (size_t)m0 * K;

  auto stage = [&](int buf, int k0) {
    const _Float16* g = gsrc_row + k0;
    _Float16* l = &s_b[buf][nrow][cq * 8];
#if HAVE_ASYNC_LDS
    __builtin_amdgcn_global_load_async_to_lds_b128(
        (as1_b128*)g, (as3_b128*)l, 0, 0);
#else
    *reinterpret_cast<uint4*>(l) = *reinterpret_cast<const uint4*>(g);
#endif
  };
  auto stage_commit = [&]() {
#if HAVE_ASYNC_LDS
    __builtin_amdgcn_s_wait_asynccnt(0);
#endif
    __syncthreads();
  };

  stage(0, 0);
  stage_commit();

  for (int k0 = 0; k0 < K; k0 += 32) {
    const int cur = (k0 >> 5) & 1;
    if (k0 + 32 < K) stage(cur ^ 1, k0 + 32);  // overlap next tile's DMA
    if (k0 + 256 < K)                          // global_prefetch_b8 on A stream
      __builtin_prefetch(a_ptr + (size_t)(lane & 15) * K + k0 + 256, 0, 1);

    v16h a = load_a16(a_ptr + k0, K, lane);
#pragma unroll
    for (int j = 0; j < 4; ++j) {
      v16h b = load_bt16(&s_b[cur][j * 16][0], BPAD, lane);
      acc[j] = WMMA_F16(a, b, acc[j]);
    }
    stage_commit();  // next buffer ready + this buffer's reads done
  }

  // C/D layout: lane col N = lane&15, rows = r + 8*(lane>=16)
  const int mbase = m0 + ((lane >> 4) << 3);
#pragma unroll
  for (int j = 0; j < 4; ++j) {
    const int c = n0 + j * 16 + (lane & 15);
    if (MODE == 2) {
      float* o = (float*)Cout;
#pragma unroll
      for (int r = 0; r < 8; ++r) o[(size_t)(mbase + r) * N + c] = acc[j][r];
    } else {
      _Float16* o = (_Float16*)Cout;
      const int bb = mbase / S_;      // tiles never straddle batch boundary
      const int s  = mbase % S_;
      const int hh = c >> 7;          // head
      const int d  = c & (HD - 1);    // dim within head
      if (MODE == 0) {
        size_t base = (((size_t)bb * H_ + hh) * S_ + s) * HD + d;
#pragma unroll
        for (int r = 0; r < 8; ++r) o[base + (size_t)r * HD] = (_Float16)acc[j][r];
      } else {  // MODE 1: [B,H,hd,S] -> 8 consecutive halfs per lane
        size_t base = (((size_t)bb * H_ + hh) * HD + d) * S_ + s;
#pragma unroll
        for (int r = 0; r < 8; ++r) o[base + r] = (_Float16)acc[j][r];
      }
    }
  }
}

// ---------------------------------------------------------------------------
// K2: block attention.  Grid (32 q-tiles, 16 heads, 2 batches); 8 waves/WG.
// Wave w: M-sub = w&3 (16 q-rows), score N-group = w>>2 (keys 0-31 / 32-63),
// PV N-group = w>>2 (hd dims 0-63 / 64-127).
// ---------------------------------------------------------------------------
__global__ __launch_bounds__(256) void block_attn(const _Float16* __restrict__ q16,
                                                  const _Float16* __restrict__ k16,
                                                  const _Float16* __restrict__ vT16,
                                                  _Float16* __restrict__ ao16,
                                                  float* __restrict__ attn) {
  const int qt = blockIdx.x, h = blockIdx.y, b = blockIdx.z;
  const int lane = threadIdx.x & 31;
  const int wave = threadIdx.x >> 5;
  const int q0 = qt * 64;
  const size_t bh = (size_t)b * H_ + h;

  const _Float16* Q  = q16  + (bh * S_ + q0) * HD;  // [64][128] row-major
  const _Float16* Kk = k16  + bh * S_ * HD;         // [2048][128] row-major
  const _Float16* Vt = vT16 + bh * HD * S_;         // [128][2048] (hd-major)
  float* attn_bh = attn + (bh * S_ + q0) * S_;

  __shared__ float    s_sc[64][65];   // padded: conflict-free
  __shared__ float    s_red[64][8];   // [0..3]=max partials, [4..7]=sum partials
  __shared__ _Float16 s_p[64][72];    // padded: 36 dwords/row, gcd(36,64)=4

  const int msub = wave & 3;
  const int ngrp = wave >> 2;

  // Q fragments live in registers for all 32 k-blocks (4 chunks of K=32)
  v16h qf[4];
#pragma unroll
  for (int c = 0; c < 4; ++c)
    qf[c] = load_a16(Q + (size_t)(msub * 16) * HD + c * 32, HD, lane);

  v8f oacc[4] = {};                           // hd cols: ngrp*64 + j*16
  const float rscale = 0.08838834764831845f;  // 1/sqrt(128)

  for (int kb = 0; kb < NB; ++kb) {
    // ---- scores: S_tile = Q(64x128) @ K_blk^T ----
    const _Float16* Kb = Kk + (size_t)(kb * 64) * HD;  // [64 keys][128]
    v8f sc[2] = {};
#pragma unroll
    for (int c = 0; c < 4; ++c) {
#pragma unroll
      for (int j = 0; j < 2; ++j) {
        v16h bf = load_bt16(Kb + (size_t)((ngrp * 2 + j) * 16) * HD + c * 32, HD, lane);
        sc[j] = WMMA_F16(qf[c], bf, sc[j]);
      }
    }
#pragma unroll
    for (int j = 0; j < 2; ++j) {
      const int col = (ngrp * 2 + j) * 16 + (lane & 15);
      const int rb = msub * 16 + ((lane >> 4) << 3);
#pragma unroll
      for (int r = 0; r < 8; ++r) s_sc[rb + r][col] = sc[j][r] * rscale;
    }
    __syncthreads();

    // ---- per-block softmax: 4 threads per row, 16 cols each ----
    {
      const int row = threadIdx.x >> 2;
      const int qq = threadIdx.x & 3;
      float mx = -3.0e38f;
#pragma unroll
      for (int c = 0; c < 16; ++c) mx = fmaxf(mx, s_sc[row][qq * 16 + c]);
      s_red[row][qq] = mx;
      __syncthreads();
      const float M4 = fmaxf(fmaxf(s_red[row][0], s_red[row][1]),
                             fmaxf(s_red[row][2], s_red[row][3]));
      float e[16], sum = 0.f;
#pragma unroll
      for (int c = 0; c < 16; ++c) {
        e[c] = __expf(s_sc[row][qq * 16 + c] - M4);
        sum += e[c];
      }
      s_red[row][4 + qq] = sum;
      __syncthreads();
      const float inv =
          1.0f / (s_red[row][4] + s_red[row][5] + s_red[row][6] + s_red[row][7]);
      float* arow = attn_bh + (size_t)row * S_ + kb * 64 + qq * 16;
#pragma unroll
      for (int c = 0; c < 16; ++c) {
        const float p = e[c] * inv;
        arow[c] = p;                       // materialized attn weights (f32)
        s_p[row][qq * 16 + c] = (_Float16)p;
      }
    }
    __syncthreads();

    // ---- out += P(64x64) @ V_blk(64x128), V pre-transposed [hd][S] ----
#pragma unroll
    for (int ks = 0; ks < 2; ++ks) {
      v16h pf = load_a16(&s_p[msub * 16][0] + ks * 32, 72, lane);
#pragma unroll
      for (int j = 0; j < 4; ++j) {
        v16h vf = load_bt16(Vt + (size_t)(ngrp * 64 + j * 16) * S_ + kb * 64 + ks * 32,
                            S_, lane);
        oacc[j] = WMMA_F16(pf, vf, oacc[j]);
      }
    }
    __syncthreads();  // s_sc / s_p are rewritten next iteration
  }

  // store per-head output, reassembled as [B*S, D] f16 for the Wo GEMM
#pragma unroll
  for (int j = 0; j < 4; ++j) {
    const int dcol = h * HD + ngrp * 64 + j * 16 + (lane & 15);
    const int srow = q0 + msub * 16 + ((lane >> 4) << 3);
#pragma unroll
    for (int r = 0; r < 8; ++r)
      ao16[((size_t)b * S_ + srow + r) * D_ + dcol] = (_Float16)oacc[j][r];
  }
}

// ---------------------------------------------------------------------------
extern "C" void kernel_launch(void* const* d_in, const int* in_sizes, int n_in,
                              void* d_out, int out_size, void* d_ws, size_t ws_size,
                              hipStream_t stream) {
  const float* query = (const float*)d_in[0];
  const float* key_  = (const float*)d_in[1];
  const float* value = (const float*)d_in[2];
  const float* Wq = (const float*)d_in[3];
  const float* Wk = (const float*)d_in[4];
  const float* Wv = (const float*)d_in[5];
  const float* Wo = (const float*)d_in[6];
  // d_in[7] = block_size (=64, hardcoded)

  float* out  = (float*)d_out;                 // [B,S,D]
  float* attn = out + (size_t)B_ * S_ * D_;    // [B,H,S,S]

  const size_t XN = (size_t)B_ * S_ * D_;  // 8,388,608
  const size_t WN = (size_t)D_ * D_;       // 4,194,304

  _Float16* w    = (_Float16*)d_ws;
  _Float16* xq   = w;
  _Float16* xk   = xq + XN;
  _Float16* xv   = xk + XN;
  _Float16* wq16 = xv + XN;
  _Float16* wk16 = wq16 + WN;
  _Float16* wv16 = wk16 + WN;
  _Float16* wo16 = wv16 + WN;
  _Float16* q16  = wo16 + WN;
  _Float16* k16  = q16 + XN;
  _Float16* vT16 = k16 + XN;
  _Float16* ao16 = vT16 + XN;  // total ~151 MB of ws

  auto cvt = [&](const float* s, _Float16* d, size_t n) {
    int grid = (int)((n / 4 + 255) / 256);
    if (grid > 8192) grid = 8192;
    cvt_f32_f16<<<grid, 256, 0, stream>>>(s, d, n);
  };
  cvt(query, xq, XN);
  cvt(key_,  xk, XN);
  cvt(value, xv, XN);
  cvt(Wq, wq16, WN);
  cvt(Wk, wk16, WN);
  cvt(Wv, wv16, WN);
  cvt(Wo, wo16, WN);

  const int M = B_ * S_;  // 4096
  dim3 ggrid(M / 128, D_ / 64);  // 32 x 32
  gemm_rt<0><<<ggrid, 256, 0, stream>>>(xq, wq16, q16,  M, D_, D_);
  gemm_rt<0><<<ggrid, 256, 0, stream>>>(xk, wk16, k16,  M, D_, D_);
  gemm_rt<1><<<ggrid, 256, 0, stream>>>(xv, wv16, vT16, M, D_, D_);

  block_attn<<<dim3(S_ / 64, H_, B_), 256, 0, stream>>>(q16, k16, vT16, ao16, attn);

  gemm_rt<2><<<ggrid, 256, 0, stream>>>(ao16, wo16, out, M, D_, D_);
}
